// GptOssTopKRouter_13374528160266
// MI455X (gfx1250) — compile-verified, matrix operand order checked by reading, not run on hardware
//
#include <hip/hip_runtime.h>
#include <hip/hip_bf16.h>
#include <math.h>

typedef __bf16 bf16_t;
typedef __attribute__((ext_vector_type(8)))  bf16_t v8bf;
typedef __attribute__((ext_vector_type(16))) bf16_t v16bf;
typedef __attribute__((ext_vector_type(8)))  float  v8f;
typedef __attribute__((ext_vector_type(4)))  unsigned int v4u;
typedef __attribute__((ext_vector_type(4)))  int          v4i;

#define D_MODEL   2880
#define N_EXPERTS 32
#define TOPK      4
#define KCHUNK    32
#define NCHUNKS   (D_MODEL / KCHUNK)        // 90
#define WAVES_PB  8
#define TOK_TILES 2                          // token tiles per wave
#define TOK_PW    (TOK_TILES * 16)           // 32 tokens per wave
#define TOK_PB    (WAVES_PB * TOK_PW)        // 256 tokens per block

// Load one 16-bit A/B WMMA fragment (16x32 MxK striping, ISA 7.12.2):
// lanes 0-15 hold K = {0..7, 16..23}, lanes 16-31 hold K = {8..15, 24..31}.
// `p` already points at row_base + kchunk + kbase; two 16B loads.

// Regular-temporal variant: used for W (184 KB, must stay cache-resident).
__device__ inline v16bf load_frag(const bf16_t* __restrict__ p) {
  v8bf lo = *reinterpret_cast<const v8bf*>(p);        // K = kbase .. kbase+7
  v8bf hi = *reinterpret_cast<const v8bf*>(p + 16);   // K = kbase+16 .. kbase+23
  v16bf r;
#pragma unroll
  for (int i = 0; i < 8; ++i) { r[i] = lo[i]; r[i + 8] = hi[i]; }
  return r;
}

// Non-temporal variant: used for x (189 MB streamed exactly once) so the
// stream does not evict W from WGP$/L2 (CDNA5 TH=NT load hint, ISA 7.3).
__device__ inline v16bf load_frag_nt(const bf16_t* __restrict__ p) {
  v8bf lo = __builtin_nontemporal_load(reinterpret_cast<const v8bf*>(p));
  v8bf hi = __builtin_nontemporal_load(reinterpret_cast<const v8bf*>(p + 16));
  v16bf r;
#pragma unroll
  for (int i = 0; i < 8; ++i) { r[i] = lo[i]; r[i + 8] = hi[i]; }
  return r;
}

__device__ inline float round_bf16(float x) {
  return (float)(bf16_t)x;   // RNE round through bf16, matches bf16 einsum output
}

__global__ __launch_bounds__(256) void gptoss_topk_router_kernel(
    const bf16_t* __restrict__ X,      // [T, D] bf16
    const bf16_t* __restrict__ W,      // [E, D] bf16
    const bf16_t* __restrict__ Bias,   // [E]    bf16
    bf16_t* __restrict__ scores,       // [T, E] bf16
    int* __restrict__ indices,         // [T, K] int32
    int T) {
  __shared__ float sm[WAVES_PB][TOK_PW][N_EXPERTS];   // 32 KB

  const int tid   = threadIdx.x;
  const int wave  = tid >> 5;
  const int lane  = tid & 31;
  const int laneN = lane & 15;        // row (A) / expert column (B/C) within tile
  const int hiHalf = lane >> 4;       // 0/1: which K-half this lane holds
  const int kbase  = hiHalf * 8;

  const long waveTokBase = (long)blockIdx.x * TOK_PB + (long)wave * TOK_PW;

  // ---- GEMM: 32 tokens x 32 experts per wave, K streamed in 32-wide chunks ----
  const bf16_t* aPtr[TOK_TILES];
#pragma unroll
  for (int t = 0; t < TOK_TILES; ++t) {
    long row = waveTokBase + t * 16 + laneN;
    if (row > T - 1) row = T - 1;     // clamp (T is a multiple of 256 in practice)
    aPtr[t] = X + (size_t)row * D_MODEL + kbase;
  }
  const bf16_t* b0Ptr = W + (size_t)laneN * D_MODEL + kbase;          // experts 0..15
  const bf16_t* b1Ptr = W + (size_t)(laneN + 16) * D_MODEL + kbase;   // experts 16..31

  v8f acc[TOK_TILES][2];
#pragma unroll
  for (int t = 0; t < TOK_TILES; ++t)
#pragma unroll
    for (int n = 0; n < 2; ++n)
      acc[t][n] = (v8f){0.f, 0.f, 0.f, 0.f, 0.f, 0.f, 0.f, 0.f};

#pragma unroll 2
  for (int kc = 0; kc < NCHUNKS; ++kc) {
    const int off = kc * KCHUNK;
    v16bf b0 = load_frag(b0Ptr + off);      // cache-hot: W is 184 KB total
    v16bf b1 = load_frag(b1Ptr + off);
#pragma unroll
    for (int t = 0; t < TOK_TILES; ++t) {
      v16bf a = load_frag_nt(aPtr[t] + off);  // NT: x read exactly once from HBM
      acc[t][0] = __builtin_amdgcn_wmma_f32_16x16x32_bf16(
          false, a, false, b0, (short)0, acc[t][0], false, false);
      acc[t][1] = __builtin_amdgcn_wmma_f32_16x16x32_bf16(
          false, a, false, b1, (short)0, acc[t][1], false, false);
    }
  }

  // ---- bias add + bf16 rounding, spill C tiles to LDS ----
  // C layout (ISA 7.12.2): lane n<16 -> N=n, VGPR r -> M=r; lane>=16 -> N=n-16, M=r+8.
  const float bias0 = (float)Bias[laneN];
  const float bias1 = (float)Bias[laneN + 16];
#pragma unroll
  for (int t = 0; t < TOK_TILES; ++t) {
#pragma unroll
    for (int r = 0; r < 8; ++r) {
      const int m = t * 16 + r + 8 * hiHalf;     // token slot within wave
      sm[wave][m][laneN]      = round_bf16(acc[t][0][r] + bias0);
      sm[wave][m][laneN + 16] = round_bf16(acc[t][1][r] + bias1);
    }
  }
  __syncthreads();

  // ---- per-token top-4 + softmax + dense scatter (1 token per lane) ----
  const long tok = waveTokBase + lane;
  if (tok < T) {
    float v[N_EXPERTS];
#pragma unroll
    for (int e = 0; e < N_EXPERTS; ++e) v[e] = sm[wave][lane][e];

    float tv[TOPK];
    int   ti[TOPK];
#pragma unroll
    for (int k = 0; k < TOPK; ++k) {
      float best = -INFINITY;
      int   bi   = 0;
      for (int e = 0; e < N_EXPERTS; ++e) {     // strict '>' => lowest index wins ties
        if (v[e] > best) { best = v[e]; bi = e; }
      }
      tv[k] = best;
      ti[k] = bi;
      v[bi] = -INFINITY;
    }

    // softmax over the 4 selected logits (tv[0] is the max)
    float ex[TOPK], s = 0.f;
#pragma unroll
    for (int k = 0; k < TOPK; ++k) { ex[k] = __expf(tv[k] - tv[0]); s += ex[k]; }
    const float inv = 1.f / s;

    // pack 32 bf16 scores (28 zeros + 4 probs) into 16 dwords
    unsigned int pk[N_EXPERTS / 2];
#pragma unroll
    for (int i = 0; i < N_EXPERTS / 2; ++i) pk[i] = 0u;
#pragma unroll
    for (int k = 0; k < TOPK; ++k) {
      bf16_t h = (bf16_t)(ex[k] * inv);
      unsigned int bits = (unsigned int)__builtin_bit_cast(unsigned short, h);
      pk[ti[k] >> 1] |= bits << (16 * (ti[k] & 1));
    }

    // NT stores: results are write-once, never re-read by this kernel.
    v4u* srow = reinterpret_cast<v4u*>(scores + (size_t)tok * N_EXPERTS);
#pragma unroll
    for (int i = 0; i < 4; ++i) {
      v4u q = (v4u){pk[4 * i], pk[4 * i + 1], pk[4 * i + 2], pk[4 * i + 3]};
      __builtin_nontemporal_store(q, srow + i);
    }

    v4i qi = (v4i){ti[0], ti[1], ti[2], ti[3]};
    __builtin_nontemporal_store(
        qi, reinterpret_cast<v4i*>(indices + (size_t)tok * TOPK));
  }
}

extern "C" void kernel_launch(void* const* d_in, const int* in_sizes, int n_in,
                              void* d_out, int out_size, void* d_ws, size_t ws_size,
                              hipStream_t stream) {
  (void)n_in; (void)out_size; (void)d_ws; (void)ws_size;
  const bf16_t* X    = (const bf16_t*)d_in[0];   // hidden_states [T, D]
  const bf16_t* W    = (const bf16_t*)d_in[1];   // weight [E, D]
  const bf16_t* Bias = (const bf16_t*)d_in[2];   // bias [E]
  const int T = in_sizes[0] / D_MODEL;           // 32768

  bf16_t* scores = (bf16_t*)d_out;                                   // [T, E] bf16
  int* indices = (int*)((char*)d_out + (size_t)T * N_EXPERTS * 2);   // [T, K] i32

  const int blocks = (T + TOK_PB - 1) / TOK_PB;  // 128
  gptoss_topk_router_kernel<<<blocks, 256, 0, stream>>>(X, W, Bias, scores,
                                                        indices, T);
}
